// ContextAttention_52381421142031
// MI455X (gfx1250) — matrix operand order repaired and checked by reading, not compiled, not run on hardware
//
#include <hip/hip_runtime.h>

// ---------------------------------------------------------------------------
// ContextAttention for MI455X (gfx1250, wave32, WMMA, TDM).
//   out  = softmax(mask ? -inf : (inp @ (inp@Wk^T)^T / sqrt(D))) @ (inp@Wv^T)
// Bandwidth-bound on the 268MB attn write (~15us floor @ 23.3TB/s); all GEMMs
// via v_wmma_f32_16x16x32_f16. K/V/Q f16 copies (12MB) stay L2-resident.
// Scores for a 16-query tile stay resident in 256KB LDS (CDNA5: 320KB/WGP).
// Q tile staged into LDS via the Tensor Data Mover (TENSORcnt).
// ---------------------------------------------------------------------------

typedef __attribute__((ext_vector_type(16))) _Float16 v16h;
typedef __attribute__((ext_vector_type(8)))  _Float16 v8h;
typedef __attribute__((ext_vector_type(8)))  float    v8f;
typedef __attribute__((ext_vector_type(4)))  float    v4f;
typedef __attribute__((ext_vector_type(4)))  unsigned int u32x4;
typedef __attribute__((ext_vector_type(8)))  int      i32x8;
typedef __attribute__((ext_vector_type(4)))  int      i32x4;

#define DIMD   128
#define BATCH  4
#define SEQ    4096
#define SC_STRIDE (SEQ + 4)          // padded LDS row stride (floats): kills bank conflicts
#define SC_BYTES  (16u * SC_STRIDE * 4u)
#define NWAVE  8                     // 256 threads / wave32

static __device__ __forceinline__ v8f wmma16(v16h a, v16h b, v8f c) {
  // (neg_a, A, neg_b, B, c_mod, C, reuse_a, reuse_b)
  return __builtin_amdgcn_wmma_f32_16x16x32_f16(false, a, false, b, (short)0, c, false, false);
}

// A-operand (16x32 f16, M=lane&15). Lane pattern: K = h*16..+15 (contiguous).
// p points at row_base + h*16; second group at +8.
static __device__ __forceinline__ v16h load_a_h(const _Float16* p) {
  v8h g0 = *(const v8h*)p;
  v8h g1 = *(const v8h*)(p + 8);
  v16h r;
#pragma unroll
  for (int i = 0; i < 8; ++i) { r[i] = g0[i]; r[i + 8] = g1[i]; }
  return r;
}

// B-operand (32x16 f16, N=lane&15). Lane holds 16 contiguous K (K = h*16..+15).
static __device__ __forceinline__ v16h load_b_h(const _Float16* p) {
  v8h g0 = *(const v8h*)p;
  v8h g1 = *(const v8h*)(p + 8);
  v16h r;
#pragma unroll
  for (int i = 0; i < 8; ++i) { r[i] = g0[i]; r[i + 8] = g1[i]; }
  return r;
}

// A pattern, converting from f32.
static __device__ __forceinline__ v16h cvt_a_f32(const float* p) {
  v4f a0 = *(const v4f*)p,        a1 = *(const v4f*)(p + 4);
  v4f b0 = *(const v4f*)(p + 8), b1 = *(const v4f*)(p + 12);
  v16h r;
#pragma unroll
  for (int i = 0; i < 4; ++i) {
    r[i]      = (_Float16)a0[i];  r[i + 4]  = (_Float16)a1[i];
    r[i + 8]  = (_Float16)b0[i];  r[i + 12] = (_Float16)b1[i];
  }
  return r;
}

// B pattern, converting 16 contiguous f32.
static __device__ __forceinline__ v16h cvt_b_f32(const float* p) {
  v16h r;
#pragma unroll
  for (int i = 0; i < 16; i += 4) {
    v4f v = *(const v4f*)(p + i);
#pragma unroll
    for (int j = 0; j < 4; ++j) r[i + j] = (_Float16)v[j];
  }
  return r;
}

// ---------------------------------------------------------------------------
// Kernel 1: K = inp@Wk^T, V = inp@Wv^T (f16), plus Q = f16(inp).
// ---------------------------------------------------------------------------
__global__ __launch_bounds__(256) void proj_kernel(
    const float* __restrict__ inp, const float* __restrict__ wk,
    const float* __restrict__ wv,  _Float16* __restrict__ Qh,
    _Float16* __restrict__ Kh,     _Float16* __restrict__ Vt) {
  const int tid  = threadIdx.x;
  const int lane = tid & 31, wave = tid >> 5;
  const int g0   = blockIdx.x * 16;
  const int batch = g0 / SEQ;
  const int srow  = g0 - batch * SEQ;
  const int nlan = lane & 15, hsel = lane >> 4;

  {  // f16 copy of inp tile, coalesced b128 -> b128
    size_t base = (size_t)g0 * DIMD + (size_t)tid * 8;
    v4f x0 = *(const v4f*)(inp + base);
    v4f x1 = *(const v4f*)(inp + base + 4);
    v8h h;
#pragma unroll
    for (int i = 0; i < 4; ++i) { h[i] = (_Float16)x0[i]; h[i + 4] = (_Float16)x1[i]; }
    *(v8h*)(Qh + base) = h;
  }

  v16h A[4];
#pragma unroll
  for (int c = 0; c < 4; ++c)
    A[c] = cvt_a_f32(inp + (size_t)(g0 + nlan) * DIMD + c * 32 + hsel * 16);

  const int e0 = wave * 16;
  const int ecol = e0 + nlan;
  v8f ck = {}, cv = {};
#pragma unroll
  for (int c = 0; c < 4; ++c) {
    v16h bk = cvt_b_f32(wk + (size_t)ecol * DIMD + c * 32 + hsel * 16);
    ck = wmma16(A[c], bk, ck);
    v16h bv = cvt_b_f32(wv + (size_t)ecol * DIMD + c * 32 + hsel * 16);
    cv = wmma16(A[c], bv, cv);
  }

#pragma unroll
  for (int r = 0; r < 8; ++r) {   // C layout: (m = r + 8*hsel, n = nlan)
    int m = r + 8 * hsel;
    Kh[(size_t)(g0 + m) * DIMD + e0 + nlan] = (_Float16)ck[r];
    Vt[((size_t)batch * DIMD + e0 + nlan) * SEQ + srow + m] = (_Float16)cv[r];
  }
}

// ---------------------------------------------------------------------------
// Kernel 2: fused scores + softmax(+mask) + attn-write + attn@V per q-tile.
// LDS: float sc[16][SC_STRIDE] (256.25KB) + _Float16 qstage[16*128] (4KB).
// ---------------------------------------------------------------------------
__global__ __launch_bounds__(256) void attn_kernel(
    const unsigned char* __restrict__ mask, const _Float16* __restrict__ Qh,
    const _Float16* __restrict__ Kh, const _Float16* __restrict__ Vt,
    float* __restrict__ out, float* __restrict__ attn) {
  extern __shared__ char smem_raw[];
  float*    sc = (float*)smem_raw;                    // [16][SC_STRIDE]
  _Float16* qs = (_Float16*)(smem_raw + SC_BYTES);    // [16][128] Q tile

  const int tid  = threadIdx.x;
  const int lane = tid & 31, wave = tid >> 5;
  const int g0   = blockIdx.x * 16;
  const int batch = g0 / SEQ;
  const int q0    = g0 - batch * SEQ;
  const int nlan = lane & 15, hsel = lane >> 4;

  // ---- stage Q tile (4KB) into LDS via Tensor Data Mover ----
  const _Float16* Qg = Qh + (size_t)g0 * DIMD;
#if __has_builtin(__builtin_amdgcn_tensor_load_to_lds)
  if (wave == 0) {
    unsigned long long ga = (unsigned long long)(uintptr_t)Qg;
    unsigned lds_off = __builtin_amdgcn_groupstaticsize() + SC_BYTES;
    // D# group0: count=1 | lds_addr | global_addr[56:0] | type=2
    u32x4 g0d = { 1u, lds_off, (unsigned)(ga & 0xffffffffu),
                  (unsigned)((ga >> 32) & 0x01ffffffu) | (2u << 30) };
    // D# group1: data_size=1 (2B); 1-D tile of 2048 elems; dim0_stride=2048
    i32x8 g1d = { (int)(1u << 16),            // workgroup_mask=0, data_size=2B
                  (int)(2048u << 16),         // tensor_dim0[15:0]=2048
                  0,                          // tensor_dim0 hi / tensor_dim1 lo=0
                  (int)(2048u << 16),         // tile_dim0=2048
                  0, 2048, 0, 0 };            // tile_dim1/2=0; dim0_stride=2048
    i32x4 g2d = {}, g3d = {};
#if defined(__clang_major__) && (__clang_major__ >= 23)
    i32x8 g4d = {};
    __builtin_amdgcn_tensor_load_to_lds(g0d, g1d, g2d, g3d, g4d, 0);
#else
    __builtin_amdgcn_tensor_load_to_lds(g0d, g1d, g2d, g3d, 0);
#endif
    __builtin_amdgcn_s_wait_tensorcnt(0);
  }
#else
  *(v8h*)(qs + tid * 8) = *(const v8h*)(Qg + tid * 8);
#endif
  __syncthreads();

  // Query A-operands from LDS, held in VGPRs for the whole score sweep
  v16h A[4];
#pragma unroll
  for (int c = 0; c < 4; ++c)
    A[c] = load_a_h(qs + nlan * DIMD + c * 32 + hsel * 16);

  const _Float16* Kb = Kh + (size_t)batch * SEQ * DIMD;
  const unsigned char* Mb = mask + (size_t)batch * SEQ * SEQ + (size_t)q0 * SEQ;
  const float scale = 0.08838834764831845f;  // 1/sqrt(128)

  // ---- phase 1: raw scores -> LDS; wave w owns k-tiles w, w+8, ...
  for (int kt = wave; kt < SEQ / 16; kt += NWAVE) {
    const int k0 = kt * 16;
    const _Float16* kp = Kb + (size_t)(k0 + nlan) * DIMD;  // B col = K row
    v8f c = {};
#pragma unroll
    for (int cc = 0; cc < 4; ++cc)
      c = wmma16(A[cc], load_b_h(kp + cc * 32 + hsel * 16), c);
    if (kt + NWAVE < SEQ / 16)
      __builtin_prefetch(Kb + (size_t)(k0 + 16 * NWAVE + nlan) * DIMD, 0, 1);
#pragma unroll
    for (int r = 0; r < 8; ++r) {
      const int m = r + 8 * hsel;
      sc[m * SC_STRIDE + k0 + nlan] = c[r] * scale;
    }
  }
  __syncthreads();

  // ---- phase 2: mask + softmax; wave w owns rows 2w, 2w+1 (no cross-wave deps)
#pragma unroll
  for (int rr = 0; rr < 2; ++rr) {
    const int m = wave * 2 + rr;
    float* row = sc + m * SC_STRIDE;
    const unsigned char* mrow = Mb + (size_t)m * SEQ;
    float mx = -__builtin_inff();
    for (int k = lane * 4; k < SEQ; k += 128) {     // coalesced u32 mask loads
      v4f v = *(const v4f*)(row + k);
      unsigned mb = *(const unsigned*)(mrow + k);
#pragma unroll
      for (int j = 0; j < 4; ++j)
        if ((mb >> (8 * j)) & 0xffu) v[j] = -__builtin_inff();
      *(v4f*)(row + k) = v;                         // masked scores back to LDS
      mx = fmaxf(mx, fmaxf(fmaxf(v[0], v[1]), fmaxf(v[2], v[3])));
    }
#pragma unroll
    for (int off = 16; off; off >>= 1) mx = fmaxf(mx, __shfl_xor(mx, off));
    float sum = 0.f;
    for (int k = lane * 4; k < SEQ; k += 128) {
      v4f v = *(const v4f*)(row + k);
      sum += __expf(v[0] - mx) + __expf(v[1] - mx) + __expf(v[2] - mx) + __expf(v[3] - mx);
    }
#pragma unroll
    for (int off = 16; off; off >>= 1) sum += __shfl_xor(sum, off);
    const float inv = 1.0f / sum;
    float* arow = attn + ((size_t)batch * SEQ + q0 + m) * SEQ;
    for (int k = lane * 4; k < SEQ; k += 128) {
      v4f v = *(const v4f*)(row + k);
      v4f p;
#pragma unroll
      for (int j = 0; j < 4; ++j) p[j] = __expf(v[j] - mx) * inv;
      *(v4f*)(row + k) = p;      // normalized probs for phase 3
      *(v4f*)(arow + k) = p;     // the big 268MB attn output write (b128, coalesced)
    }
  }
  __syncthreads();

  // ---- phase 3: out = attn @ V; wave w owns D-slab [16w, 16w+16)
  const int e0 = wave * 16;
  const _Float16* Vb = Vt + ((size_t)batch * DIMD + e0 + nlan) * SEQ;  // B col = Vt row
  v8f acc = {};
  for (int k0 = 0; k0 < SEQ; k0 += 32) {
    const float* ap = sc + nlan * SC_STRIDE + k0 + hsel * 16;  // A lane pattern
    v4f a0 = *(const v4f*)ap,        a1 = *(const v4f*)(ap + 4);
    v4f b0 = *(const v4f*)(ap + 8), b1 = *(const v4f*)(ap + 12);
    v16h a;
#pragma unroll
    for (int i = 0; i < 4; ++i) {
      a[i]      = (_Float16)a0[i];  a[i + 4]  = (_Float16)a1[i];
      a[i + 8]  = (_Float16)b0[i];  a[i + 12] = (_Float16)b1[i];
    }
    v16h b = load_b_h(Vb + k0 + hsel * 16);
    acc = wmma16(a, b, acc);
  }
#pragma unroll
  for (int r = 0; r < 8; ++r) {
    const int m = r + 8 * hsel;
    out[(size_t)(g0 + m) * DIMD + e0 + nlan] = acc[r];
  }
}

// ---------------------------------------------------------------------------
extern "C" void kernel_launch(void* const* d_in, const int* in_sizes, int n_in,
                              void* d_out, int out_size, void* d_ws, size_t ws_size,
                              hipStream_t stream) {
  const float* inp = (const float*)d_in[0];
  const float* wk  = (const float*)d_in[1];
  const float* wv  = (const float*)d_in[2];
  const unsigned char* mask = (const unsigned char*)d_in[3];  // jnp.bool_ = 1 byte

  float* out  = (float*)d_out;                          // [B,S,D]
  float* attn = out + (size_t)BATCH * SEQ * DIMD;       // [B,S,S]

  _Float16* Qh = (_Float16*)d_ws;                       // 4 MB
  _Float16* Kh = Qh + (size_t)BATCH * SEQ * DIMD;       // 4 MB
  _Float16* Vt = Kh + (size_t)BATCH * SEQ * DIMD;       // 4 MB ([B][D][S])

  const int ntiles = BATCH * SEQ / 16;                  // 1024
  proj_kernel<<<ntiles, 256, 0, stream>>>(inp, wk, wv, Qh, Kh, Vt);

  const size_t lds_bytes = SC_BYTES + 16 * DIMD * sizeof(_Float16);  // 266,496 B
  attn_kernel<<<ntiles, 256, lds_bytes, stream>>>(mask, Qh, Kh, Vt, out, attn);
}
